// DenseDistance_38817914422094
// MI455X (gfx1250) — matrix operand order repaired and checked by reading, not compiled, not run on hardware
//
#include <hip/hip_runtime.h>

typedef __attribute__((ext_vector_type(16))) __bf16 v16bf;
typedef __attribute__((ext_vector_type(8)))  float  v8f;

#define B_DIM 2048
#define D_DIM 256
#define U_DIM 512
#define EPSV  1e-7f

#define MT (B_DIM / 16)   // 128 M tiles (16 rows each)
#define NT (U_DIM / 16)   // 32  N tiles
#define KC (D_DIM / 32)   // 8   K chunks of 32

// d_ws layout (bytes). Fragments stored pre-swizzled in WMMA lane order:
// fragment f occupies 32 lanes * 32 bytes = 1 KB; lane's 16 bf16 at f*1KB + lane*32.
#define AHI_OFF  0
#define ALO_OFF  (AHI_OFF + (size_t)B_DIM * D_DIM * 2)   // 1 MB each
#define BHI_OFF  (ALO_OFF + (size_t)B_DIM * D_DIM * 2)
#define BLO_OFF  (BHI_OFF + (size_t)D_DIM * U_DIM * 2)   // 256 KB each
#define XN_OFF   (BLO_OFF + (size_t)D_DIM * U_DIM * 2)   // 2048 f32
#define WN_OFF   (XN_OFF  + (size_t)B_DIM * 4)           // 512 f32

__device__ __forceinline__ void bf16_split(float v, __bf16& hi, __bf16& lo) {
  hi = (__bf16)v;
  lo = (__bf16)(v - (float)hi);
}

// ---- Pre-pack X into A-fragments (16x32 bf16 layout) + exact row norms ----
__global__ __launch_bounds__(256)
void prepack_x(const float* __restrict__ X, unsigned char* __restrict__ ws) {
  const int lane  = threadIdx.x & 31;
  const int wave  = threadIdx.x >> 5;
  const int mTile = blockIdx.x * 8 + wave;          // 0..127
  const int m     = lane & 15;
  const int hs    = lane >> 4;

  const float* xrow  = X + (size_t)(mTile * 16 + m) * D_DIM;
  v16bf* Ahi  = (v16bf*)(ws + AHI_OFF);
  v16bf* Alo  = (v16bf*)(ws + ALO_OFF);
  float* xnrm = (float*)(ws + XN_OFF);

  float xnPart = 0.f;
  for (int kc = 0; kc < KC; ++kc) {
    // A layout: lane<16 -> K = 32kc+0..15 ; lane>=16 -> K = 32kc+16..31
    const float4* p0 = (const float4*)(xrow + kc * 32 + 16 * hs);
    const float4* p1 = (const float4*)(xrow + kc * 32 + 16 * hs + 8);
    float4 g0a = p0[0], g0b = p0[1], g1a = p1[0], g1b = p1[1];
    float af[16] = {g0a.x, g0a.y, g0a.z, g0a.w, g0b.x, g0b.y, g0b.z, g0b.w,
                    g1a.x, g1a.y, g1a.z, g1a.w, g1b.x, g1b.y, g1b.z, g1b.w};
    v16bf hi, lo;
#pragma unroll
    for (int e = 0; e < 16; ++e) {
      __bf16 h, l;
      bf16_split(af[e], h, l);
      hi[e] = h; lo[e] = l;
      xnPart = fmaf(af[e], af[e], xnPart);          // exact fp32 norm
    }
    const int frag = mTile * KC + kc;
    Ahi[frag * 32 + lane] = hi;                     // wave writes 1KB contiguous
    Alo[frag * 32 + lane] = lo;
  }
  const float xn = xnPart + __shfl_xor(xnPart, 16, 32);
  if (lane < 16) xnrm[mTile * 16 + m] = xn;
}

// ---- Pre-pack W into B-fragments (32x16 bf16 layout) + exact col norms ----
__global__ __launch_bounds__(256)
void prepack_w(const float* __restrict__ W, unsigned char* __restrict__ ws) {
  const int lane  = threadIdx.x & 31;
  const int wave  = threadIdx.x >> 5;
  const int nTile = blockIdx.x * 8 + wave;          // 0..31
  const int n     = nTile * 16 + (lane & 15);
  const int hs    = lane >> 4;

  v16bf* Bhi  = (v16bf*)(ws + BHI_OFF);
  v16bf* Blo  = (v16bf*)(ws + BLO_OFF);
  float* wnrm = (float*)(ws + WN_OFF);

  float wnPart = 0.f;
  for (int kc = 0; kc < KC; ++kc) {
    // B layout: lane<16 -> K = 32kc+0..15 ; lane>=16 -> K = 32kc+16..31
    const int kb = kc * 32 + 16 * hs;
    v16bf hi, lo;
#pragma unroll
    for (int e = 0; e < 16; ++e) {
      float v = W[(size_t)(kb + e) * U_DIM + n];    // coalesced across lanes
      __bf16 h, l;
      bf16_split(v, h, l);
      hi[e] = h; lo[e] = l;
      wnPart = fmaf(v, v, wnPart);
    }
    const int frag = nTile * KC + kc;
    Bhi[frag * 32 + lane] = hi;
    Blo[frag * 32 + lane] = lo;
  }
  const float wn = wnPart + __shfl_xor(wnPart, 16, 32);
  if (lane < 16) wnrm[n] = wn;
}

// ---- Main GEMM: 32(M) x 64(N) per wave, conversion-free bf16x3 WMMA loop ----
// sq = ||x||^2 + ||w||^2 - 2*(x.w); out = sqrt(max(sq, eps))
__global__ __launch_bounds__(256)
void dense_l2_wmma(const unsigned char* __restrict__ ws, float* __restrict__ Out) {
  const int lane    = threadIdx.x & 31;
  const int wave    = threadIdx.x >> 5;
  const int mTile0  = blockIdx.x * 2;               // two 16-row tiles per wave
  const int rowBase = blockIdx.x * 32;
  const int nTile0  = wave * 4;                     // 4 N tiles per wave
  const int m       = lane & 15;
  const int hs      = lane >> 4;

  const v16bf* Ahi  = (const v16bf*)(ws + AHI_OFF);
  const v16bf* Alo  = (const v16bf*)(ws + ALO_OFF);
  const v16bf* Bhi  = (const v16bf*)(ws + BHI_OFF);
  const v16bf* Blo  = (const v16bf*)(ws + BLO_OFF);
  const float* xnrm = (const float*)(ws + XN_OFF);
  const float* wnrm = (const float*)(ws + WN_OFF);

  v8f acc[2][4];
#pragma unroll
  for (int s = 0; s < 2; ++s)
#pragma unroll
    for (int t = 0; t < 4; ++t) acc[s][t] = v8f{};

#pragma unroll 2   // cap unroll: 64 acc VGPRs live; avoid spills from over-prefetch
  for (int kc = 0; kc < KC; ++kc) {
    const int a0 = (mTile0 * KC + kc) * 32 + lane;
    const int a1 = ((mTile0 + 1) * KC + kc) * 32 + lane;
    const v16bf aHi0 = Ahi[a0], aLo0 = Alo[a0];     // 2x b128 each, no conversion
    const v16bf aHi1 = Ahi[a1], aLo1 = Alo[a1];
#pragma unroll
    for (int t = 0; t < 4; ++t) {
      const int bfr = ((nTile0 + t) * KC + kc) * 32 + lane;
      const v16bf bHi = Bhi[bfr];
      const v16bf bLo = Blo[bfr];
      // bf16x3: hi*hi + hi*lo + lo*hi recovers ~fp32 dot; B reused across both M tiles
      acc[0][t] = __builtin_amdgcn_wmma_f32_16x16x32_bf16(
          false, aHi0, false, bHi, (short)0, acc[0][t], false, false);
      acc[0][t] = __builtin_amdgcn_wmma_f32_16x16x32_bf16(
          false, aHi0, false, bLo, (short)0, acc[0][t], false, false);
      acc[0][t] = __builtin_amdgcn_wmma_f32_16x16x32_bf16(
          false, aLo0, false, bHi, (short)0, acc[0][t], false, false);
      acc[1][t] = __builtin_amdgcn_wmma_f32_16x16x32_bf16(
          false, aHi1, false, bHi, (short)0, acc[1][t], false, false);
      acc[1][t] = __builtin_amdgcn_wmma_f32_16x16x32_bf16(
          false, aHi1, false, bLo, (short)0, acc[1][t], false, false);
      acc[1][t] = __builtin_amdgcn_wmma_f32_16x16x32_bf16(
          false, aLo1, false, bHi, (short)0, acc[1][t], false, false);
    }
  }

#pragma unroll
  for (int s = 0; s < 2; ++s) {
    const float xn_m = xnrm[rowBase + s * 16 + m];  // lane m holds row m's norm
#pragma unroll
    for (int t = 0; t < 4; ++t) {
      const int   col = (nTile0 + t) * 16 + m;
      const float wn  = wnrm[col];
#pragma unroll
      for (int r = 0; r < 8; ++r) {
        const int   mr  = r + 8 * hs;               // C/D layout row for vgpr r
        const float xnr = __shfl(xn_m, mr, 32);
        const float sq  = fmaf(-2.f, acc[s][t][r], xnr + wn);
        Out[(size_t)(rowBase + s * 16 + mr) * U_DIM + col] =
            __builtin_sqrtf(fmaxf(sq, EPSV));       // coalesced 64B per half-wave
      }
    }
  }
}

extern "C" void kernel_launch(void* const* d_in, const int* in_sizes, int n_in,
                              void* d_out, int out_size, void* d_ws, size_t ws_size,
                              hipStream_t stream) {
  (void)in_sizes; (void)n_in; (void)out_size; (void)ws_size;
  const float* x  = (const float*)d_in[0];   // (2048, 256) fp32
  const float* w  = (const float*)d_in[1];   // (256, 512)  fp32
  float*       o  = (float*)d_out;           // (2048, 512) fp32
  unsigned char* ws = (unsigned char*)d_ws;  // ~2.6 MB used

  prepack_x<<<dim3(MT / 8), dim3(256), 0, stream>>>(x, ws);     // 128 waves
  prepack_w<<<dim3(NT / 8), dim3(256), 0, stream>>>(w, ws);     // 32 waves
  dense_l2_wmma<<<dim3(MT / 2), dim3(256), 0, stream>>>(ws, o); // 512 waves, 32x64 each
}
